// GCNMultiLabel_21904333209751
// MI455X (gfx1250) — compile-verified, matrix operand order checked by reading, not run on hardware
//
#include <hip/hip_runtime.h>
#include <stdint.h>

typedef __attribute__((ext_vector_type(16))) __bf16 v16bf;
typedef __attribute__((ext_vector_type(8)))  float  v8f;

// ---------- helpers ----------
__device__ __forceinline__ uint32_t pk_bf16(float a, float b) {
    // round-to-nearest-even f32 -> bf16, packed pair
    uint32_t ua = __float_as_uint(a), ub = __float_as_uint(b);
    ua = (ua + 0x7FFFu + ((ua >> 16) & 1u)) >> 16;
    ub = (ub + 0x7FFFu + ((ub >> 16) & 1u)) >> 16;
    return (ua & 0xFFFFu) | (ub << 16);
}

__device__ __forceinline__ uint16_t bf16r(float f) {
    uint32_t u = __float_as_uint(f);
    u = (u + 0x7FFFu + ((u >> 16) & 1u)) >> 16;
    return (uint16_t)u;
}

struct U32x8 { uint4 a, b; };          // 32B -> bit_cast to v16bf

// ---------- degree / norm ----------
__global__ void __launch_bounds__(256) k_fill1(float* p, int n) {
    int i = blockIdx.x * 256 + threadIdx.x;
    if (i < n) p[i] = 1.0f;            // self-loop contributes 1 to degree
}

__global__ void __launch_bounds__(256) k_deg_edges(const int* __restrict__ dst,
                                                   float* __restrict__ deg, int E) {
    int i = blockIdx.x * 256 + threadIdx.x;
    if (i < E)
        __hip_atomic_fetch_add(&deg[dst[i]], 1.0f, __ATOMIC_RELAXED,
                               __HIP_MEMORY_SCOPE_AGENT);
}

__global__ void __launch_bounds__(256) k_rsqrt_ip(float* p, int n) {
    int i = blockIdx.x * 256 + threadIdx.x;
    if (i < n) p[i] = rsqrtf(p[i]);
}

// ---------- pack weight matrix into WMMA B-fragment layout (bf16) ----------
// Flat layout: word[((nt*KC + kc)*32 + lane)*8 + v] holds B[k][col],B[k+1][col]
// with col = nt*16 + (lane&15), k = kc*32 + (v>>2)*16 + (lane>>4)*8 + (v&3)*2.
__global__ void __launch_bounds__(256) k_pack_w(const float* __restrict__ W,
                                                uint32_t* __restrict__ out,
                                                int K, int N, int NT) {
    int idx = blockIdx.x * 256 + threadIdx.x;
    int KC = K >> 5;
    int total = NT * KC * 256;
    if (idx >= total) return;
    int v    = idx & 7;
    int lane = (idx >> 3) & 31;
    int rest = idx >> 8;
    int kc   = rest % KC;
    int nt   = rest / KC;
    int col  = nt * 16 + (lane & 15);
    int k    = kc * 32 + ((v >> 2) << 4) + ((lane >> 4) << 3) + ((v & 3) << 1);
    float f0 = 0.f, f1 = 0.f;
    if (col < N) {
        f0 = W[(size_t)k * N + col];
        f1 = W[(size_t)(k + 1) * N + col];
    }
    out[idx] = pk_bf16(f0, f1);
}

// ---------- WMMA GEMM: C[M,Nout] = act(A[M,K] (+aBias, relu)) * B (+oBias) ----
// One wave per 16-row M tile; NT 16-col N tiles accumulated per wave.
// OBF16: store C as bf16 (intermediate xw buffers); else f32.
template <int NT, bool AREL, bool OBIAS, bool OBF16>
__global__ void __launch_bounds__(256)
k_gemm_wmma(const float* __restrict__ A, const float* __restrict__ aBias,
            const uint32_t* __restrict__ Bp, const float* __restrict__ oBias,
            void* __restrict__ Cv, int M, int K, int Nout) {
    const int lane  = threadIdx.x & 31;
    const int wave  = threadIdx.x >> 5;
    const int mTile = blockIdx.x * 8 + wave;
    if (mTile * 16 >= M) return;                 // whole-wave exit: EXEC stays all-ones
    const int KC   = K >> 5;
    const int half = lane >> 4;
    const int l15  = lane & 15;
    const int rowA = mTile * 16 + l15;           // A: lanes 0-15 & 16-31 both M=0..15

    v8f acc[NT];
#pragma unroll
    for (int nt = 0; nt < NT; ++nt)
#pragma unroll
        for (int v = 0; v < 8; ++v) acc[nt][v] = 0.0f;

    for (int kc = 0; kc < KC; ++kc) {
        // A fragment: 16x32 bf16, per CDNA5 ISA layout
        struct { uint32_t w[8]; } araw;
#pragma unroll
        for (int v = 0; v < 8; ++v) {
            int k = kc * 32 + ((v >> 2) << 4) + (half << 3) + ((v & 3) << 1);
            float2 av = *(const float2*)(A + (size_t)rowA * K + k);
            float f0 = av.x, f1 = av.y;
            if (AREL) {
                f0 = fmaxf(f0 + aBias[k], 0.0f);
                f1 = fmaxf(f1 + aBias[k + 1], 0.0f);
            }
            araw.w[v] = pk_bf16(f0, f1);
        }
        v16bf afrag = __builtin_bit_cast(v16bf, araw);

        // issue ALL B-fragment loads first so they clause together, then WMMAs
        U32x8 braw[NT];
#pragma unroll
        for (int nt = 0; nt < NT; ++nt) {
            const uint4* p = (const uint4*)(Bp + ((size_t)(nt * KC + kc) * 32 + lane) * 8);
            braw[nt].a = p[0];
            braw[nt].b = p[1];
        }
#pragma unroll
        for (int nt = 0; nt < NT; ++nt) {
            v16bf bfrag = __builtin_bit_cast(v16bf, braw[nt]);
            acc[nt] = __builtin_amdgcn_wmma_f32_16x16x32_bf16(
                false, afrag, false, bfrag, (short)0, acc[nt], false, false);
        }
    }
    // C/D layout: n = lane&15, m = v + 8*(lane>>4)
#pragma unroll
    for (int nt = 0; nt < NT; ++nt) {
        int col = nt * 16 + l15;
        if (col < Nout) {
            if (OBF16) {
                uint16_t* C16 = (uint16_t*)Cv;
#pragma unroll
                for (int v = 0; v < 8; ++v) {
                    int row = mTile * 16 + v + 8 * half;
                    C16[(size_t)row * Nout + col] = bf16r(acc[nt][v]);
                }
            } else {
                float* C = (float*)Cv;
#pragma unroll
                for (int v = 0; v < 8; ++v) {
                    int row = mTile * 16 + v + 8 * half;
                    float val = acc[nt][v];
                    if (OBIAS) val += oBias[col];
                    C[(size_t)row * Nout + col] = val;
                }
            }
        }
    }
}

// ---------- self-loop init: agg[i,:] = xw[i,:] * dinv[i]^2 (xw is bf16) ------
__global__ void __launch_bounds__(256) k_self_init(const uint16_t* __restrict__ xw16,
                                                   const float* __restrict__ dinv,
                                                   float* __restrict__ agg, int n32) {
    int p = blockIdx.x * 256 + threadIdx.x;
    if (p >= n32) return;
    int node = p >> 5;
    int j    = p & 31;
    float d  = dinv[node];
    float s  = d * d;
    uint32_t u = *(const uint32_t*)(xw16 + (size_t)node * 64 + j * 2);
    float2 o;
    o.x = __uint_as_float(u << 16) * s;
    o.y = __uint_as_float(u & 0xFFFF0000u) * s;
    ((float2*)(agg + (size_t)node * 64))[j] = o;
}

// ---------- edge scatter: agg[dst,:] += xw[src,:] * dinv[src]*dinv[dst] ------
// one wave per edge, 2 bf16 features per lane (64 features), f32 atomic accum
__global__ void __launch_bounds__(256) k_edge_agg(const int* __restrict__ src,
                                                  const int* __restrict__ dst,
                                                  const float* __restrict__ dinv,
                                                  const uint16_t* __restrict__ xw16,
                                                  float* __restrict__ agg, int E) {
    int e = blockIdx.x * 8 + (threadIdx.x >> 5);
    if (e >= E) return;
    int lane = threadIdx.x & 31;
    int s = src[e];
    int d = dst[e];
    float w = dinv[s] * dinv[d];
    uint32_t u = *(const uint32_t*)(xw16 + (size_t)s * 64 + lane * 2);
    float f0 = __uint_as_float(u << 16);
    float f1 = __uint_as_float(u & 0xFFFF0000u);
    float* out = agg + (size_t)d * 64 + lane * 2;
    __hip_atomic_fetch_add(out,     f0 * w, __ATOMIC_RELAXED, __HIP_MEMORY_SCOPE_AGENT);
    __hip_atomic_fetch_add(out + 1, f1 * w, __ATOMIC_RELAXED, __HIP_MEMORY_SCOPE_AGENT);
}

// ---------------------------------------------------------------------------
extern "C" void kernel_launch(void* const* d_in, const int* in_sizes, int n_in,
                              void* d_out, int out_size, void* d_ws, size_t ws_size,
                              hipStream_t stream) {
    const float* x    = (const float*)d_in[0];
    const int*   ei   = (const int*)d_in[1];
    const float* W1   = (const float*)d_in[2];
    const float* b1   = (const float*)d_in[3];
    const float* W2   = (const float*)d_in[4];
    const float* b2   = (const float*)d_in[5];
    const float* Wout = (const float*)d_in[6];
    const float* bout = (const float*)d_in[7];

    const int H    = in_sizes[3];            // 64
    const int IN   = in_sizes[2] / H;        // 128
    const int N    = in_sizes[0] / IN;       // 100000
    const int E    = in_sizes[1] / 2;        // 3200000
    const int NLAB = in_sizes[7];            // 121

    // workspace carve-up (256B aligned)
    char* ws = (char*)d_ws;
    size_t off = 0;
    auto take = [&](size_t bytes) -> char* {
        size_t r = (off + 255) & ~(size_t)255;
        off = r + bytes;
        return ws + r;
    };
    float*     dinv = (float*)take((size_t)N * 4);
    uint16_t*  xw   = (uint16_t*)take((size_t)N * H * 2);   // bf16 intermediate
    float*     agg  = (float*)take((size_t)N * H * 4);
    uint32_t*  bp   = (uint32_t*)take(65536);

    const int* srcI = ei;
    const int* dstI = ei + E;

    const int gN   = (N + 255) / 256;
    const int gE   = (E + 255) / 256;
    const int gEw  = (E + 7) / 8;            // 1 wave / edge
    const int gM   = (N / 16 + 7) / 8;       // 1 wave / 16-row tile
    const int gN32 = ((N * 32) + 255) / 256;

    // degrees -> dinv (in place)
    k_fill1<<<gN, 256, 0, stream>>>(dinv, N);
    k_deg_edges<<<gE, 256, 0, stream>>>(dstI, dinv, E);
    k_rsqrt_ip<<<gN, 256, 0, stream>>>(dinv, N);

    // ---- layer 1: xw = x @ W1 (bf16 out) ; agg = scatter(norm * xw) ----
    {
        int KC = IN >> 5, NT = H / 16;
        k_pack_w<<<(NT * KC * 256 + 255) / 256, 256, 0, stream>>>(W1, bp, IN, H, NT);
    }
    k_gemm_wmma<4, false, false, true><<<gM, 256, 0, stream>>>(
        x, nullptr, bp, nullptr, (void*)xw, N, IN, H);
    k_self_init<<<gN32, 256, 0, stream>>>(xw, dinv, agg, N * 32);
    k_edge_agg<<<gEw, 256, 0, stream>>>(srcI, dstI, dinv, xw, agg, E);

    // ---- layer 2: xw2 = relu(agg + b1) @ W2 (bf16 out) ; agg2 = scatter ----
    {
        int KC = H >> 5, NT = H / 16;
        k_pack_w<<<(NT * KC * 256 + 255) / 256, 256, 0, stream>>>(W2, bp, H, H, NT);
    }
    k_gemm_wmma<4, true, false, true><<<gM, 256, 0, stream>>>(
        agg, b1, bp, nullptr, (void*)xw, N, H, H);
    k_self_init<<<gN32, 256, 0, stream>>>(xw, dinv, agg, N * 32);
    k_edge_agg<<<gEw, 256, 0, stream>>>(srcI, dstI, dinv, xw, agg, E);

    // ---- output: logits = relu(agg + b2) @ Wout + bout (f32 to d_out) ----
    {
        int KC = H >> 5, NT = 8;             // 8*16 = 128 >= 121, cols guarded
        k_pack_w<<<(NT * KC * 256 + 255) / 256, 256, 0, stream>>>(Wout, bp, H, NLAB, NT);
    }
    k_gemm_wmma<8, true, true, false><<<gM, 256, 0, stream>>>(
        agg, b2, bp, bout, d_out, N, H, NLAB);
}
